// Conv2D_QuantModule_73856257622363
// MI455X (gfx1250) — matrix operand order repaired and checked by reading, not compiled, not run on hardware
//
#include <hip/hip_runtime.h>

typedef _Float16 f16;
typedef __attribute__((ext_vector_type(16))) _Float16 v16h;
typedef __attribute__((ext_vector_type(8)))  _Float16 v8h;
typedef __attribute__((ext_vector_type(8)))  float     v8f;

#define B_N   32
#define C_IN  128
#define C_OUT 256
#define H_    56
#define W_    56
#define HW_   (H_ * W_)
#define NW_EL (C_OUT * C_IN * 9)
#define LDT   40   // LDS row stride in f16 elements (80B: 16B aligned, bank-spread)

// ---------------- workspace init ----------------
__global__ void init_ws_kernel(unsigned int* m) { *m = 0u; }

// ---------------- max |w| reduction ----------------
__global__ __launch_bounds__(256) void maxabs_kernel(const float* __restrict__ w,
                                                     int n, unsigned int* __restrict__ out) {
    __shared__ float red[256];
    float m = 0.0f;
    for (int i = blockIdx.x * blockDim.x + threadIdx.x; i < n; i += gridDim.x * blockDim.x)
        m = fmaxf(m, fabsf(w[i]));
    red[threadIdx.x] = m;
    __syncthreads();
    for (int s = 128; s > 0; s >>= 1) {
        if ((int)threadIdx.x < s)
            red[threadIdx.x] = fmaxf(red[threadIdx.x], red[threadIdx.x + s]);
        __syncthreads();
    }
    if (threadIdx.x == 0)
        atomicMax(out, __float_as_uint(red[0]));  // valid: values are non-negative
}

// ---------------- ternary quantize -> f16, layout [tap][co][ci] ----------------
__global__ __launch_bounds__(256) void quant_kernel(const float* __restrict__ w,
                                                    const float* __restrict__ Wn,
                                                    const unsigned int* __restrict__ maxbits,
                                                    f16* __restrict__ qw) {
    const int i = blockIdx.x * 256 + threadIdx.x;   // i < NW_EL exactly
    const float maxv = __uint_as_float(*maxbits);
    const float nw = w[i] / maxv;
    const float t = 0.05f;
    float m = (nw > -t && nw <= t) ? 0.0f : nw;     // faithful to reference masking
    if (m > t)  m = 1.0f;
    if (m < -t) m = -1.0f;
    if (m == -1.0f) m = Wn[0];                      // Wn = -1.0
    // source layout [co][ci][kh][kw]
    const int kw = i % 3;
    int tmp = i / 3;
    const int kh = tmp % 3; tmp /= 3;
    const int ci = tmp % C_IN;
    const int co = tmp / C_IN;
    const int p = kh * 3 + kw;
    qw[(p * C_OUT + co) * C_IN + ci] = (f16)m;
}

// ---------------- X: NCHW f32 -> NHWC f16 (LDS tile transpose) ----------------
__global__ __launch_bounds__(256) void x_nhwc_f16_kernel(const float* __restrict__ X,
                                                         f16* __restrict__ Xh) {
    __shared__ float t[32][33];
    const int hwb = blockIdx.x * 32;   // 98 blocks (3136/32)
    const int cib = blockIdx.y * 32;   // 4 blocks
    const int n   = blockIdx.z;        // 32
    const int c = threadIdx.x & 31;
    const int r = threadIdx.x >> 5;    // 0..7
    #pragma unroll
    for (int i = 0; i < 4; ++i) {
        const int ci = cib + r + i * 8;
        t[r + i * 8][c] = X[(n * C_IN + ci) * HW_ + hwb + c];   // coalesced along hw
    }
    __syncthreads();
    #pragma unroll
    for (int i = 0; i < 4; ++i) {
        const int hw = hwb + r + i * 8;
        Xh[(hw + n * HW_) * C_IN + cib + c] = (f16)t[c][r + i * 8];  // coalesced along ci
    }
}

// ---------------- implicit-GEMM conv, f16 WMMA, async-to-LDS double buffer ----------------
__global__ __launch_bounds__(256) void conv_wmma_kernel(const f16* __restrict__ Xh,
                                                        const f16* __restrict__ qw,
                                                        float* __restrict__ out) {
    __shared__ __attribute__((aligned(16))) f16 At[2][128 * LDT];  // weights [co][ci]
    __shared__ __attribute__((aligned(16))) f16 Bt[2][128 * LDT];  // acts    [s][ci]

    const int tid  = threadIdx.x;
    const int lane = tid & 31;
    const int wv   = tid >> 5;        // 8 waves
    const int wm   = wv & 3;          // 4 along M
    const int wn   = wv >> 2;         // 2 along N
    const int hf   = lane >> 4;
    const int l15  = lane & 15;
    const int nb = blockIdx.x;        // 784 spatial blocks
    const int mb = blockIdx.y;        // 2 C_out blocks

    // staging geometry: this thread owns 16B chunks at rows r0, r0+64, column kc
    const int r0 = tid >> 2;          // 0..63
    const int kc = (tid & 3) * 8;     // 0,8,16,24 (f16 elements)

    // spatial decode for the two B rows this thread stages
    int bh[2], bw[2], bnb[2];
    #pragma unroll
    for (int i = 0; i < 2; ++i) {
        const int s = nb * 128 + r0 + i * 64;
        const int n = s / HW_;
        const int hw = s - n * HW_;
        bh[i]  = hw / W_;
        bw[i]  = hw - bh[i] * W_;
        bnb[i] = n * HW_;
    }
    const unsigned abase = (unsigned)((mb * 128 + r0) * C_IN + kc) * 2u;  // byte off in qw

    // issue async direct-to-LDS loads for one K-step into buffer `nbuf`
    auto stage_async = [&](int nbuf, int step) {
        const int p = step >> 2, cb = step & 3;
        const int kh = p / 3, kw = p - kh * 3;

        // --- A tile rows r0, r0+64 (always valid) ---
        const unsigned aoff0 = abase + (unsigned)(p * (C_OUT * C_IN) + cb * 32) * 2u;
        const unsigned aoff1 = aoff0 + (unsigned)(64 * C_IN) * 2u;
        const unsigned lA0 = (unsigned)(uintptr_t)&At[nbuf][r0 * LDT + kc];
        const unsigned lA1 = (unsigned)(uintptr_t)&At[nbuf][(r0 + 64) * LDT + kc];
        asm volatile("global_load_async_to_lds_b128 %0, %1, %2"
                     :: "v"(lA0), "v"(aoff0), "s"(qw) : "memory");
        asm volatile("global_load_async_to_lds_b128 %0, %1, %2"
                     :: "v"(lA1), "v"(aoff1), "s"(qw) : "memory");

        // --- B tile rows r0, r0+64 (zero-pad invalid rows with a plain DS store) ---
        #pragma unroll
        for (int i = 0; i < 2; ++i) {
            const int ih = bh[i] + kh - 1, iw = bw[i] + kw - 1;
            const bool valid = ((unsigned)ih < (unsigned)H_) && ((unsigned)iw < (unsigned)W_);
            f16* ldst = &Bt[nbuf][(r0 + i * 64) * LDT + kc];
            if (valid) {
                const unsigned boff =
                    (unsigned)((bnb[i] + ih * W_ + iw) * C_IN + cb * 32 + kc) * 2u;
                const unsigned lB = (unsigned)(uintptr_t)ldst;
                asm volatile("global_load_async_to_lds_b128 %0, %1, %2"
                             :: "v"(lB), "v"(boff), "s"(Xh) : "memory");
            } else {
                v8h z;
                #pragma unroll
                for (int e = 0; e < 8; ++e) z[e] = (f16)0.0f;
                *(v8h*)ldst = z;
            }
        }
    };

    v8f acc[2][4];
    #pragma unroll
    for (int i = 0; i < 2; ++i)
        #pragma unroll
        for (int j = 0; j < 4; ++j)
            #pragma unroll
            for (int e = 0; e < 8; ++e)
                acc[i][j][e] = 0.0f;

    // prologue: fill buffer 0
    stage_async(0, 0);
    asm volatile("s_wait_asynccnt 0x0" ::: "memory");

    for (int step = 0; step < 36; ++step) {       // 9 taps x 4 ci-blocks
        const int buf = step & 1;
        __syncthreads();                           // buf fully resident, buf^1 free

        if (step + 1 < 36) stage_async(buf ^ 1, step + 1);   // DMA overlaps WMMAs

        // --- fragments per ISA 7.12.2 layouts ---
        v16h a[2], b[4];
        #pragma unroll
        for (int fm = 0; fm < 2; ++fm) {
            const f16* rp = &At[buf][(wm * 32 + fm * 16 + l15) * LDT + hf * 8];
            const v8h lo = *(const v8h*)rp;        // K = hf*8 .. +7
            const v8h hi = *(const v8h*)(rp + 16); // K = 16+hf*8 .. +7
            #pragma unroll
            for (int e = 0; e < 8; ++e) { a[fm][e] = lo[e]; a[fm][e + 8] = hi[e]; }
        }
        #pragma unroll
        for (int fn = 0; fn < 4; ++fn) {
            const f16* cp = &Bt[buf][(wn * 64 + fn * 16 + l15) * LDT + hf * 8];
            const v8h lo = *(const v8h*)cp;        // K = hf*8 .. +7
            const v8h hi = *(const v8h*)(cp + 16); // K = 16+hf*8 .. +7
            #pragma unroll
            for (int e = 0; e < 8; ++e) { b[fn][e] = lo[e]; b[fn][e + 8] = hi[e]; }
        }

        #pragma unroll
        for (int fm = 0; fm < 2; ++fm)
            #pragma unroll
            for (int fn = 0; fn < 4; ++fn)
                acc[fm][fn] = __builtin_amdgcn_wmma_f32_16x16x32_f16(
                    false, a[fm], false, b[fn], (short)0, acc[fm][fn], false, false);

        if (step + 1 < 36)
            asm volatile("s_wait_asynccnt 0x0" ::: "memory");  // next buf landed before barrier
    }

    // --- writeback: C/D layout -> NCHW fp32 ---
    #pragma unroll
    for (int fn = 0; fn < 4; ++fn) {
        const int so  = nb * 128 + wn * 64 + fn * 16 + l15;
        const int n2  = so / HW_;
        const int hw2 = so - n2 * HW_;
        #pragma unroll
        for (int fm = 0; fm < 2; ++fm) {
            const int co = mb * 128 + wm * 32 + fm * 16 + hf * 8;
            float* dst = out + (n2 * C_OUT + co) * HW_ + hw2;
            #pragma unroll
            for (int j = 0; j < 8; ++j)
                dst[j * HW_] = acc[fm][fn][j];
        }
    }
}

// ---------------- launcher ----------------
extern "C" void kernel_launch(void* const* d_in, const int* in_sizes, int n_in,
                              void* d_out, int out_size, void* d_ws, size_t ws_size,
                              hipStream_t stream) {
    const float* X   = (const float*)d_in[0];
    const float* wgt = (const float*)d_in[1];
    // d_in[2] = Wp (unused by the faithful reference)
    const float* Wn  = (const float*)d_in[3];
    float* out = (float*)d_out;

    // workspace layout: [0] maxbits, [256B] qw (576KB), [1MB] X-f16 NHWC (25.7MB)
    unsigned int* maxbits = (unsigned int*)d_ws;
    f16* qw = (f16*)((char*)d_ws + 256);
    f16* Xh = (f16*)((char*)d_ws + (1u << 20));

    init_ws_kernel<<<1, 1, 0, stream>>>(maxbits);
    maxabs_kernel<<<288, 256, 0, stream>>>(wgt, NW_EL, maxbits);
    quant_kernel<<<NW_EL / 256, 256, 0, stream>>>(wgt, Wn, maxbits, qw);
    x_nhwc_f16_kernel<<<dim3(98, 4, 32), 256, 0, stream>>>(X, Xh);

    dim3 grid(784, 2);   // 784 spatial blocks of 128, 2 C_out blocks of 128
    conv_wmma_kernel<<<grid, 256, 0, stream>>>(Xh, qw, out);
}
